// LSTMReset_59742995087669
// MI455X (gfx1250) — matrix operand order repaired and checked by reading, not compiled, not run on hardware
//
#include <hip/hip_runtime.h>

// ---------------- problem constants ----------------
#define TT 64        // bptt
#define HH 512       // hidden
#define G4 2048      // 4*H gate columns
#define KK 1024      // concat K = [x|h]
#define NKT 32       // K tiles of 32
#define NG 16        // row-groups (16 rows each)
#define SP 4         // workgroups cooperating per group (split of N)
#define RW 16        // batch rows per group
#define WGS 256      // 8 waves of 32
#define LHALVES (G4 * KK)   // packed halves per layer = 2,097,152

typedef __attribute__((ext_vector_type(16))) __bf16 v16bf;
typedef __attribute__((ext_vector_type(8)))  float  v8f;

// ---------------- helpers ----------------
__device__ __forceinline__ unsigned short f2bf(float f) {
  unsigned int u = __float_as_uint(f);
  u += 0x7FFFu + ((u >> 16) & 1u);          // round-to-nearest-even
  return (unsigned short)(u >> 16);
}
__device__ __forceinline__ unsigned int pack2bf(float lo, float hi) {
  return (unsigned int)f2bf(lo) | ((unsigned int)f2bf(hi) << 16);
}
__device__ __forceinline__ float sigf(float x) { return 1.0f / (1.0f + __expf(-x)); }
__device__ __forceinline__ float tanh_fast(float x) {
  x = fminf(fmaxf(x, -15.0f), 15.0f);
  float e = __expf(2.0f * x);
  return (e - 1.0f) / (e + 1.0f);
}

// monotonic split barrier over L2 atomics (one counter per group)
__device__ __forceinline__ void bar_arrive(int* c) {
  __hip_atomic_fetch_add(c, 1, __ATOMIC_RELEASE, __HIP_MEMORY_SCOPE_AGENT);
}
__device__ __forceinline__ void bar_wait(int* c, int target) {
  while (__hip_atomic_load(c, __ATOMIC_ACQUIRE, __HIP_MEMORY_SCOPE_AGENT) < target)
    __builtin_amdgcn_s_sleep(2);
}

// Store one bf16 pair (even K) into the WMMA-A-fragment-tiled LDS buffer.
// A fragment layout (16-bit A 16x32): lane = row + 16*((kk>>3)&1),
// half-index idx = (kk&7) | ((kk>>4)<<3).  Tile t stored at uints [(t*32+lane)*8 ..].
__device__ __forceinline__ void stageA_pair(unsigned int* Abuf, int rr, int k, unsigned int val) {
  int kt  = k >> 5;
  int kk  = k & 31;
  int ln  = rr + (((kk >> 3) & 1) << 4);
  int idx = (kk & 7) | ((kk >> 4) << 3);
  Abuf[(((kt << 5) + ln) << 3) + (idx >> 1)] = val;
}

// 4 gate tiles (i,f,g,o) x 32 K-tiles of v_wmma_f32_16x16x32_bf16.
// Bb already offset to this (wave, layer): + lane*16 halves; gate stride = 32 tiles * 512 halves.
__device__ __forceinline__ void gemm4(const unsigned short* __restrict__ Bb,
                                      const unsigned short* Ash, int lane, v8f acc[4]) {
#pragma unroll 4
  for (int kt = 0; kt < NKT; ++kt) {
    v16bf a = *(const v16bf*)(Ash + (kt * 32 + lane) * 16);
#pragma unroll
    for (int gt = 0; gt < 4; ++gt) {
      v16bf b = *(const v16bf*)(Bb + gt * 16384 + kt * 512);
      acc[gt] = __builtin_amdgcn_wmma_f32_16x16x32_bf16(false, a, false, b,
                                                        (short)0, acc[gt], false, false);
    }
  }
}

// ---------------- pre-pass kernels ----------------
__global__ void zero_cnt_kernel(int* c) {
  if (threadIdx.x < NG * 16) c[threadIdx.x] = 0;
}

__global__ void pack_bias_kernel(const float* __restrict__ bi0, const float* __restrict__ bh0,
                                 const float* __restrict__ bi1, const float* __restrict__ bh1,
                                 float* __restrict__ bias) {
  int i = blockIdx.x * blockDim.x + threadIdx.x;
  if (i < 2 * G4) {
    int l = i >> 11, j = i & (G4 - 1);
    bias[i] = (l ? bi1[j] : bi0[j]) + (l ? bh1[j] : bh0[j]);
  }
}

// Pack [Wi | Wh] (f32, row-major 2048x512 each) into bf16 WMMA B-fragment tiles:
// tile (nt,kt): 32 lanes x 16 contiguous halves; K=kt*32+8*(lane>>4)+(idx&7)+16*(idx>>3).
__global__ void pack_w_kernel(const float* __restrict__ Wi0, const float* __restrict__ Wh0,
                              const float* __restrict__ Wi1, const float* __restrict__ Wh1,
                              unsigned short* __restrict__ wp) {
  unsigned i = blockIdx.x * blockDim.x + threadIdx.x;   // 0 .. 2*LHALVES-1
  int layer  = i >> 21;
  unsigned r = i & (LHALVES - 1);
  int idx  = r & 15;
  int lane = (r >> 4) & 31;
  int tl   = r >> 9;            // nt*32 + kt
  int kt   = tl & 31;
  int nt   = tl >> 5;           // 0..127
  int j    = nt * 16 + (lane & 15);
  int K    = kt * 32 + ((idx >> 3) << 4) + ((lane >> 4) << 3) + (idx & 7);
  const float* Wi = layer ? Wi1 : Wi0;
  const float* Wh = layer ? Wh1 : Wh0;
  float v = (K < HH) ? Wi[j * HH + K] : Wh[j * HH + (K - HH)];
  wp[i] = f2bf(v);
}

// ---------------- persistent recurrent kernel ----------------
// Grid: NG*SP = 64 WGs. Group g owns rows [g*16,g*16+16); slot s owns h cols [s*128,(s+1)*128).
// h0/h1 exchanged through double-buffered bf16 L2 buffers; 2 global barriers per step:
//   arrivals: init(SP) then per step A_t (after h0 write) and B_t (after h1 write).
//   single wait per step at target SP*(2+2t) subsumes all earlier phases (monotonic counter).
__global__ __launch_bounds__(WGS, 1) void lstm_persistent(
    const float* __restrict__ latent, const float* __restrict__ lstm_h,
    const float* __restrict__ lstm_c, const int* __restrict__ dones,
    const int* __restrict__ truncs, const unsigned short* __restrict__ wpack,
    const float* __restrict__ bias, unsigned short* hb0, unsigned short* hb1,
    int* counters, float* __restrict__ out) {
  __shared__ unsigned int A0u[RW * KK / 2];   // 32KB, A-fragment-tiled [x_t | h0]
  __shared__ unsigned int A1u[RW * KK / 2];   // 32KB, A-fragment-tiled [h0_new | h1]

  const int tid  = threadIdx.x;
  const int lane = tid & 31;
  const int wv   = tid >> 5;                  // wave 0..7
  const int g    = blockIdx.x / SP;
  const int s    = blockIdx.x % SP;
  const int row0 = g * RW;
  const int hcb  = s * 128 + wv * 16;         // this wave's h-column base
  const int col  = hcb + (lane & 15);
  const int hi8  = (lane >> 4) * 8;           // C/D row offset for lanes 16..31
  int* cnt = counters + g * 16;

  unsigned int* hb0u = (unsigned int*)hb0;
  unsigned int* hb1u = (unsigned int*)hb1;

  // biases (i,f,g,o) for this column, both layers
  float b0[4], b1[4];
#pragma unroll
  for (int gt = 0; gt < 4; ++gt) {
    b0[gt] = bias[gt * HH + col];
    b1[gt] = bias[G4 + gt * HH + col];
  }
  // per-wave packed-weight bases (n_tile base = s*8+wv)
  const unsigned short* B0 = wpack + (size_t)(s * 8 + wv) * NKT * 512 + lane * 16;
  const unsigned short* B1 = B0 + LHALVES;

  const v8f vzero = {0.f, 0.f, 0.f, 0.f, 0.f, 0.f, 0.f, 0.f};

  // ---- init c state (registers, persistent over all 64 steps) ----
  v8f c0v, c1v;
#pragma unroll
  for (int v = 0; v < 8; ++v) {
    int gr = row0 + v + hi8;
    c0v[v] = lstm_c[(gr * TT) * (2 * HH) + col];
    c1v[v] = lstm_c[(gr * TT) * (2 * HH) + HH + col];
  }
  // ---- init h state slices -> parity-1 exchange buffers ----
  for (int p = tid; p < RW * 64; p += WGS) {
    int rr = p >> 6;
    int k  = s * 128 + ((p & 63) << 1);
    const float* hp = lstm_h + ((row0 + rr) * TT) * (2 * HH);
    size_t o = ((size_t)((NG + g) * RW + rr) * HH + k) >> 1;
    hb0u[o] = pack2bf(hp[k], hp[k + 1]);
    hb1u[o] = pack2bf(hp[HH + k], hp[HH + k + 1]);
  }
  __threadfence();
  __syncthreads();
  if (tid == 0) bar_arrive(cnt);

  for (int t = 0; t < TT; ++t) {
    const int rp = (t + 1) & 1;               // parity holding h_{t-1}
    const int wp = t & 1;                     // parity we write h_t into

    // uniform reset mask for our 16 rows (scalarizes to s_loads)
    unsigned fm = 0;
#pragma unroll
    for (int r = 0; r < RW; ++r)
      fm |= ((dones[(row0 + r) * TT + t] | truncs[(row0 + r) * TT + t]) ? 1u : 0u) << r;

    // stage x_t (f32 -> bf16) into A0 K[0,512)
    for (int p = tid; p < RW * 256; p += WGS) {
      int rr = p >> 8;
      int k  = (p & 255) << 1;
      const float* xp = latent + ((row0 + rr) * TT + t) * HH + k;
      stageA_pair(A0u, rr, k, pack2bf(xp[0], xp[1]));
    }
    if (t == 0) {                             // wait for everyone's init h writes
      if (tid == 0) bar_wait(cnt, SP);
      __syncthreads();
      __threadfence();
    }
    // stage h0_{t-1} into A0 K[512,1024), zeroed where reset
    for (int p = tid; p < RW * 256; p += WGS) {
      int rr = p >> 8;
      int k  = (p & 255) << 1;
      unsigned int hv = hb0u[((size_t)((rp * NG + g) * RW + rr) * HH + k) >> 1];
      if ((fm >> rr) & 1u) hv = 0u;
      stageA_pair(A0u, rr, HH + k, hv);
    }
#pragma unroll
    for (int v = 0; v < 8; ++v)
      if ((fm >> (v + hi8)) & 1u) { c0v[v] = 0.0f; c1v[v] = 0.0f; }
    __syncthreads();

    // ---- layer 0: gates = [x|h0] @ [Wi0|Wh0]^T ----
    v8f acc[4];
#pragma unroll
    for (int gt = 0; gt < 4; ++gt) acc[gt] = vzero;
    gemm4(B0, (const unsigned short*)A0u, lane, acc);

#pragma unroll
    for (int v = 0; v < 8; ++v) {
      float gi = acc[0][v] + b0[0], gf = acc[1][v] + b0[1];
      float gg = acc[2][v] + b0[2], go = acc[3][v] + b0[3];
      float cn = sigf(gf) * c0v[v] + sigf(gi) * tanh_fast(gg);
      float hn = sigf(go) * tanh_fast(cn);
      c0v[v] = cn;
      hb0[(size_t)((wp * NG + g) * RW + v + hi8) * HH + col] = f2bf(hn);
    }
    __threadfence();
    __syncthreads();
    if (tid == 0) {
      bar_arrive(cnt);                        // A_t
      bar_wait(cnt, SP * (2 + 2 * t));        // all h0_t written (subsumes B_{t-1})
    }
    __syncthreads();
    __threadfence();

    // stage A1 = [h0_t | h1_{t-1}(reset)]
    for (int p = tid; p < RW * 256; p += WGS) {
      int rr = p >> 8;
      int k  = (p & 255) << 1;
      unsigned int h0v = hb0u[((size_t)((wp * NG + g) * RW + rr) * HH + k) >> 1];
      stageA_pair(A1u, rr, k, h0v);
      unsigned int h1v = hb1u[((size_t)((rp * NG + g) * RW + rr) * HH + k) >> 1];
      if ((fm >> rr) & 1u) h1v = 0u;
      stageA_pair(A1u, rr, HH + k, h1v);
    }
    __syncthreads();

    // ---- layer 1 ----
#pragma unroll
    for (int gt = 0; gt < 4; ++gt) acc[gt] = vzero;
    gemm4(B1, (const unsigned short*)A1u, lane, acc);

#pragma unroll
    for (int v = 0; v < 8; ++v) {
      float gi = acc[0][v] + b1[0], gf = acc[1][v] + b1[1];
      float gg = acc[2][v] + b1[2], go = acc[3][v] + b1[3];
      float cn = sigf(gf) * c1v[v] + sigf(gi) * tanh_fast(gg);
      float hn = sigf(go) * tanh_fast(cn);
      c1v[v] = cn;
      int gr = row0 + v + hi8;
      out[(size_t)(gr * TT + t) * HH + col] = hn;   // f32 output
      hb1[(size_t)((wp * NG + g) * RW + v + hi8) * HH + col] = f2bf(hn);
    }
    __threadfence();
    __syncthreads();
    if (tid == 0) bar_arrive(cnt);            // B_t
  }
}

// ---------------- host launcher ----------------
extern "C" void kernel_launch(void* const* d_in, const int* in_sizes, int n_in,
                              void* d_out, int out_size, void* d_ws, size_t ws_size,
                              hipStream_t stream) {
  const float* latent = (const float*)d_in[0];
  const float* lstm_h = (const float*)d_in[1];
  const float* lstm_c = (const float*)d_in[2];
  const int*   dones  = (const int*)d_in[3];
  const int*   truncs = (const int*)d_in[4];
  const float* W_ih0  = (const float*)d_in[5];
  const float* W_hh0  = (const float*)d_in[6];
  const float* b_ih0  = (const float*)d_in[7];
  const float* b_hh0  = (const float*)d_in[8];
  const float* W_ih1  = (const float*)d_in[9];
  const float* W_hh1  = (const float*)d_in[10];
  const float* b_ih1  = (const float*)d_in[11];
  const float* b_hh1  = (const float*)d_in[12];
  float* out = (float*)d_out;

  char* ws = (char*)d_ws;
  int*            counters = (int*)ws;                                  // 4 KB
  float*          bias     = (float*)(ws + 4096);                       // 16 KB
  unsigned short* wpack    = (unsigned short*)(ws + 4096 + 16384);      // 8 MB
  unsigned short* hb0      = (unsigned short*)(ws + 4096 + 16384 + (size_t)2 * LHALVES * 2);
  unsigned short* hb1      = hb0 + (size_t)2 * NG * RW * HH;            // 512 KB each pair half

  zero_cnt_kernel<<<1, 256, 0, stream>>>(counters);
  pack_bias_kernel<<<(2 * G4 + 255) / 256, 256, 0, stream>>>(b_ih0, b_hh0, b_ih1, b_hh1, bias);
  pack_w_kernel<<<(2 * LHALVES) / 256, 256, 0, stream>>>(W_ih0, W_hh0, W_ih1, W_hh1, wpack);
  lstm_persistent<<<NG * SP, WGS, 0, stream>>>(latent, lstm_h, lstm_c, dones, truncs,
                                               wpack, bias, hb0, hb1, counters, out);
}
